// EnhancedFPSNPSUltrasSOMNetwork_68247030333811
// MI455X (gfx1250) — matrix operand order repaired and checked by reading, not compiled, not run on hardware
//
#include <hip/hip_runtime.h>
#include <hip/hip_bf16.h>

// ---------------------------------------------------------------------------
// Bidirectional Mamba block for MI455X (gfx1250, wave32, WMMA).
//
// All GEMMs run on bf16 operands (pre-converted once; producers emit bf16
// copies in their epilogues), f32 accumulate via v_wmma_f32_16x16x32_bf16.
// Each wave computes a 16x32 C strip (two N tiles sharing one A fragment):
// inner loop = 3x global_load_b128-class loads -> 2x v_wmma, no conversions.
//
// Pipeline:
//   0)  one-time f32->bf16 casts of x and the 6 weight matrices
//   1)  xp    = x @ W_in_bi^T + b_in_bi              (WMMA 2048x1024x512)
//   2)  xboth = [x_f ; reverse(x_b)] (written bf16)  (permute)
//   3)  xz    = xboth @ W_in^T + b_in                (WMMA 4096x2048x512)
//   4)  xc    = silu(causal conv4 + cb)  (f32 + bf16 dual write)
//   5)  xdbl  = xc @ W_x^T               (WMMA 4096x160x1024, f32+bf16 write)
//   6)  dtp   = softplus(xdbl[:,:32] @ W_dt^T + b_dt) (WMMA 4096x1024x32)
//   7)  y     = selective scan (h[64], A[64] in VGPRs; B_t/C_t via LDS)
//   8)  y_bf  = (y + xc*D) * silu(z)     (bf16 write)
//   9)  mo    = y @ W_out^T + b_out      (WMMA 4096x512x1024)
//  10)  comb  = fwd + reverse(bwd)       (bf16 write)
//  11)  out   = comb @ W_out_bi^T + b_out_bi  (WMMA 2048x512x512)
// ---------------------------------------------------------------------------

typedef __attribute__((ext_vector_type(16))) __bf16 v16bf;
typedef __attribute__((ext_vector_type(8)))  __bf16 v8bf;
typedef __attribute__((ext_vector_type(8)))  float  v8f;

#define D_STATE 64
#define D_INNER 1024
#define SEQ     1024
#define B4      4

__device__ __forceinline__ float fsilu(float v) {
  return v / (1.0f + __expf(-v));
}
__device__ __forceinline__ float fsoftplus(float v) {
  return (v > 20.0f) ? v : __logf(1.0f + __expf(v));
}

// ---------------------------------------------------------------------------
// C(M,N) = A(M,K) @ B(N,K)^T [+bias] with A,B in bf16, C in f32.
// Block = 128 threads = 4 waves; wave computes a 16(M) x 32(N) strip:
// one A fragment feeds two v_wmma per K-step.
// Requires M%64==0, N%32==0, K%32==0 (true for every call).
// EPI: 0 = +bias; 1 = softplus(+bias); 2 = +bias and also write bf16 C2.
// ---------------------------------------------------------------------------
template <int EPI>
__global__ void __launch_bounds__(128)
gemm_bf16_wmma(const __bf16* __restrict__ A, int lda,
               const __bf16* __restrict__ B, int ldb,
               const float* __restrict__ bias,
               float* __restrict__ C, __bf16* __restrict__ C2,
               int ldc, int K) {
  const int lane  = threadIdx.x & 31;
  const int wave  = threadIdx.x >> 5;
  const int half  = lane >> 4;   // 0 or 1
  const int l16   = lane & 15;
  const int nBase = blockIdx.x * 32;            // two 16-wide N tiles
  const int mBase = blockIdx.y * 64 + wave * 16;

  const __bf16* __restrict__ arow  = A + (size_t)(mBase + l16) * lda;
  const __bf16* __restrict__ brow0 = B + (size_t)(nBase + l16) * ldb;
  const __bf16* __restrict__ brow1 = B + (size_t)(nBase + 16 + l16) * ldb;

  v8f acc0 = {};
  v8f acc1 = {};
  for (int k0 = 0; k0 < K; k0 += 32) {
    // A fragment (16x32 bf16), contiguous k order: elem i -> K=half*16+i
    const v8bf a0 = *reinterpret_cast<const v8bf*>(arow + k0 + half * 16);
    const v8bf a1 = *reinterpret_cast<const v8bf*>(arow + k0 + half * 16 + 8);
    const v16bf av = __builtin_shufflevector(a0, a1,
        0, 1, 2, 3, 4, 5, 6, 7, 8, 9, 10, 11, 12, 13, 14, 15);
    // B fragment (32x16 bf16), same k order as A: lane holds column n, elem i -> K=half*16+i
    const v16bf bv0 = *reinterpret_cast<const v16bf*>(brow0 + k0 + half * 16);
    const v16bf bv1 = *reinterpret_cast<const v16bf*>(brow1 + k0 + half * 16);

    if (k0 + 32 < K) {  // next K tile -> global_prefetch_b8
      __builtin_prefetch(arow + k0 + 32, 0, 1);
      __builtin_prefetch(brow0 + k0 + 32, 0, 1);
      __builtin_prefetch(brow1 + k0 + 32, 0, 1);
    }
    acc0 = __builtin_amdgcn_wmma_f32_16x16x32_bf16(
        false, av, false, bv0, (short)0, acc0, false, false);
    acc1 = __builtin_amdgcn_wmma_f32_16x16x32_bf16(
        false, av, false, bv1, (short)0, acc1, false, false);
  }

  const float bn0 = bias ? bias[nBase + l16] : 0.0f;
  const float bn1 = bias ? bias[nBase + 16 + l16] : 0.0f;
  // ISA C/D layout (16x16 f32): VGPR j -> M = half*8 + j, N = l16
#pragma unroll
  for (int j = 0; j < 8; ++j) {
    const size_t m = (size_t)(mBase + half * 8 + j);
    float v0 = acc0[j] + bn0;
    float v1 = acc1[j] + bn1;
    if (EPI == 1) { v0 = fsoftplus(v0); v1 = fsoftplus(v1); }
    const size_t o0 = m * ldc + nBase + l16;
    const size_t o1 = m * ldc + nBase + 16 + l16;
    C[o0] = v0;
    C[o1] = v1;
    if (EPI == 2) { C2[o0] = (__bf16)v0; C2[o1] = (__bf16)v1; }
  }
}

// One-time f32 -> bf16 cast.
__global__ void cast_bf16_kernel(const float* __restrict__ src,
                                 __bf16* __restrict__ dst) {
  const int i = blockIdx.x * blockDim.x + threadIdx.x;
  dst[i] = (__bf16)src[i];
}

// xboth(4,SEQ,512) in bf16: streams 0,1 = xp[:, :, :512];
// streams 2,3 = time-reversed xp[:, :, 512:1024].
__global__ void build_xboth_kernel(const float* __restrict__ xp,
                                   __bf16* __restrict__ xboth) {
  const int idx = blockIdx.x * blockDim.x + threadIdx.x;  // B4*SEQ*512
  const int c  = idx & 511;
  const int t  = (idx >> 9) & (SEQ - 1);
  const int b4 = idx >> 19;
  float v;
  if (b4 < 2) v = xp[((size_t)(b4 * SEQ + t)) * 1024 + c];
  else        v = xp[((size_t)((b4 - 2) * SEQ + (SEQ - 1 - t))) * 1024 + 512 + c];
  xboth[idx] = (__bf16)v;
}

// xc = silu(causal depthwise conv4 of xz[:, :, :D_INNER] + conv_b); dual write.
__global__ void conv_silu_kernel(const float* __restrict__ xz,   // (B4*SEQ,2048)
                                 const float* __restrict__ cw,   // (1024,4)
                                 const float* __restrict__ cb,   // (1024)
                                 float* __restrict__ xc,         // (B4*SEQ,1024)
                                 __bf16* __restrict__ xc_bf) {
  const int idx = blockIdx.x * blockDim.x + threadIdx.x;  // B4*SEQ*1024
  const int d = idx & (D_INNER - 1);
  const int t = (idx >> 10) & (SEQ - 1);
  const int b = idx >> 20;
  float acc = cb[d];
#pragma unroll
  for (int j = 0; j < 4; ++j) {
    const int ti = t + j - 3;
    if (ti >= 0)
      acc += cw[d * 4 + j] * xz[((size_t)(b * SEQ + ti)) * (2 * D_INNER) + d];
  }
  const float v = fsilu(acc);
  xc[idx] = v;
  xc_bf[idx] = (__bf16)v;
}

// Selective scan: one thread per (stream, d_inner) channel; h[64], A[64] in
// VGPRs; shared B_t/C_t broadcast via LDS; dt already softplus'd.
__global__ void __launch_bounds__(256)
mamba_scan_kernel(const float* __restrict__ xdbl,   // (B4*SEQ,160)
                  const float* __restrict__ dtp,    // (B4*SEQ,1024)
                  const float* __restrict__ xc,     // (B4*SEQ,1024)
                  const float* __restrict__ A_log,  // (1024,64)
                  float* __restrict__ y) {          // (B4*SEQ,1024)
  const int b = blockIdx.x >> 2;
  const int d = ((blockIdx.x & 3) << 8) + threadIdx.x;
  __shared__ float sB[D_STATE];
  __shared__ float sC[D_STATE];

  float h[D_STATE];
  float Ad[D_STATE];
#pragma unroll
  for (int s = 0; s < D_STATE; ++s) {
    h[s]  = 0.0f;
    Ad[s] = -__expf(A_log[(size_t)d * D_STATE + s]);
  }

  for (int t = 0; t < SEQ; ++t) {
    const size_t row = (size_t)b * SEQ + t;
    __syncthreads();
    if (threadIdx.x < 64)       sB[threadIdx.x]      = xdbl[row * 160 + 32 + threadIdx.x];
    else if (threadIdx.x < 128) sC[threadIdx.x - 64] = xdbl[row * 160 + 96 + (threadIdx.x - 64)];
    __syncthreads();

    const float dt = dtp[row * D_INNER + d];
    const float dx = dt * xc[row * D_INNER + d];
    float acc = 0.0f;
#pragma unroll
    for (int s = 0; s < D_STATE; ++s) {
      h[s] = __expf(dt * Ad[s]) * h[s] + dx * sB[s];
      acc += h[s] * sC[s];
    }
    y[row * D_INNER + d] = acc;
  }
}

// y_bf = (y + xc*D_param) * silu(z);  z = xz[:, :, D_INNER:]
__global__ void gate_kernel(const float* __restrict__ y,
                            const float* __restrict__ xc,
                            const float* __restrict__ xz,
                            const float* __restrict__ Dp,
                            __bf16* __restrict__ y_bf) {
  const int idx = blockIdx.x * blockDim.x + threadIdx.x;  // B4*SEQ*1024
  const int d = idx & (D_INNER - 1);
  const size_t row = (size_t)(idx >> 10);
  const float z = xz[row * (2 * D_INNER) + D_INNER + d];
  y_bf[idx] = (__bf16)((y[idx] + xc[idx] * Dp[d]) * fsilu(z));
}

// comb_bf = mo[0:2] + reverse-in-time(mo[2:4])
__global__ void combine_kernel(const float* __restrict__ mo,   // (B4*SEQ,512)
                               __bf16* __restrict__ comb) {    // (2*SEQ,512)
  const int idx = blockIdx.x * blockDim.x + threadIdx.x;  // 2*SEQ*512
  const int c = idx & 511;
  const int t = (idx >> 9) & (SEQ - 1);
  const int b = idx >> 19;
  comb[idx] = (__bf16)(mo[((size_t)(b * SEQ + t)) * 512 + c] +
                       mo[((size_t)((b + 2) * SEQ + (SEQ - 1 - t))) * 512 + c]);
}

extern "C" void kernel_launch(void* const* d_in, const int* in_sizes, int n_in,
                              void* d_out, int out_size, void* d_ws, size_t ws_size,
                              hipStream_t stream) {
  (void)in_sizes; (void)n_in; (void)out_size; (void)ws_size;
  const float* x        = (const float*)d_in[0];   // (2,1024,512)
  const float* W_in_bi  = (const float*)d_in[1];   // (1024,512)
  const float* b_in_bi  = (const float*)d_in[2];   // (1024)
  const float* W_out_bi = (const float*)d_in[3];   // (512,512)
  const float* b_out_bi = (const float*)d_in[4];   // (512)
  const float* W_in     = (const float*)d_in[5];   // (2048,512)
  const float* b_in     = (const float*)d_in[6];   // (2048)
  const float* conv_w   = (const float*)d_in[7];   // (1024,1,4)
  const float* conv_b   = (const float*)d_in[8];   // (1024)
  const float* W_x      = (const float*)d_in[9];   // (160,1024)
  const float* W_dt     = (const float*)d_in[10];  // (1024,32)
  const float* b_dt     = (const float*)d_in[11];  // (1024)
  const float* A_log    = (const float*)d_in[12];  // (1024,64)
  const float* D_param  = (const float*)d_in[13];  // (1024)
  const float* W_out    = (const float*)d_in[14];  // (512,1024)
  const float* b_out    = (const float*)d_in[15];  // (512)
  float* out = (float*)d_out;                      // (2,1024,512)

  // ---- workspace layout ----
  float* w = (float*)d_ws;
  float* xp    = w; w += (size_t)2048 * 1024;   // f32
  float* xz    = w; w += (size_t)4096 * 2048;
  float* xc    = w; w += (size_t)4096 * 1024;
  float* xdbl  = w; w += (size_t)4096 * 160;
  float* dtp   = w; w += (size_t)4096 * 1024;
  float* yb    = w; w += (size_t)4096 * 1024;
  float* mo    = w; w += (size_t)4096 * 512;
  __bf16* h = (__bf16*)w;                        // bf16 region
  __bf16* x_bf     = h; h += (size_t)2048 * 512;
  __bf16* xboth_bf = h; h += (size_t)4096 * 512;
  __bf16* xc_bf    = h; h += (size_t)4096 * 1024;
  __bf16* xdbl_bf  = h; h += (size_t)4096 * 160;
  __bf16* y_bf     = h; h += (size_t)4096 * 1024;
  __bf16* comb_bf  = h; h += (size_t)2048 * 512;
  __bf16* Wbi_bf   = h; h += (size_t)1024 * 512;
  __bf16* Win_bf   = h; h += (size_t)2048 * 512;
  __bf16* Wx_bf    = h; h += (size_t)160 * 1024;
  __bf16* Wdt_bf   = h; h += (size_t)1024 * 32;
  __bf16* Wo_bf    = h; h += (size_t)512 * 1024;
  __bf16* Wobi_bf  = h; h += (size_t)512 * 512;

  const dim3 blk(128);
  // ---- 0) one-time bf16 casts (input + weights) ----
  cast_bf16_kernel<<<(2048 * 512) / 256, 256, 0, stream>>>(x, x_bf);
  cast_bf16_kernel<<<(1024 * 512) / 256, 256, 0, stream>>>(W_in_bi, Wbi_bf);
  cast_bf16_kernel<<<(2048 * 512) / 256, 256, 0, stream>>>(W_in, Win_bf);
  cast_bf16_kernel<<<(160 * 1024) / 256, 256, 0, stream>>>(W_x, Wx_bf);
  cast_bf16_kernel<<<(1024 * 32) / 256, 256, 0, stream>>>(W_dt, Wdt_bf);
  cast_bf16_kernel<<<(512 * 1024) / 256, 256, 0, stream>>>(W_out, Wo_bf);
  cast_bf16_kernel<<<(512 * 512) / 256, 256, 0, stream>>>(W_out_bi, Wobi_bf);

  // 1) xp = x @ W_in_bi^T + b_in_bi   (M=2048, N=1024, K=512)
  gemm_bf16_wmma<0><<<dim3(1024 / 32, 2048 / 64), blk, 0, stream>>>(
      x_bf, 512, Wbi_bf, 512, b_in_bi, xp, nullptr, 1024, 512);

  // 2) 4-stream (fwd + time-reversed) input, bf16
  build_xboth_kernel<<<(B4 * SEQ * 512) / 256, 256, 0, stream>>>(xp, xboth_bf);

  // 3) xz = xboth @ W_in^T + b_in     (M=4096, N=2048, K=512)
  gemm_bf16_wmma<0><<<dim3(2048 / 32, 4096 / 64), blk, 0, stream>>>(
      xboth_bf, 512, Win_bf, 512, b_in, xz, nullptr, 2048, 512);

  // 4) depthwise causal conv + silu (f32 + bf16)
  conv_silu_kernel<<<(B4 * SEQ * D_INNER) / 256, 256, 0, stream>>>(
      xz, conv_w, conv_b, xc, xc_bf);

  // 5) xdbl = xc @ W_x^T              (M=4096, N=160, K=1024), f32 + bf16 out
  gemm_bf16_wmma<2><<<dim3(160 / 32, 4096 / 64), blk, 0, stream>>>(
      xc_bf, 1024, Wx_bf, 1024, nullptr, xdbl, xdbl_bf, 160, 1024);

  // 6) dtp = softplus(xdbl[:, :32] @ W_dt^T + b_dt)  (M=4096, N=1024, K=32)
  gemm_bf16_wmma<1><<<dim3(1024 / 32, 4096 / 64), blk, 0, stream>>>(
      xdbl_bf, 160, Wdt_bf, 32, b_dt, dtp, nullptr, 1024, 32);

  // 7) selective scan
  mamba_scan_kernel<<<16, 256, 0, stream>>>(xdbl, dtp, xc, A_log, yb);

  // 8) gate -> bf16
  gate_kernel<<<(B4 * SEQ * D_INNER) / 256, 256, 0, stream>>>(
      yb, xc, xz, D_param, y_bf);

  // 9) mo = y @ W_out^T + b_out       (M=4096, N=512, K=1024)
  gemm_bf16_wmma<0><<<dim3(512 / 32, 4096 / 64), blk, 0, stream>>>(
      y_bf, 1024, Wo_bf, 1024, b_out, mo, nullptr, 512, 1024);

  // 10) comb = fwd + reversed bwd -> bf16
  combine_kernel<<<(2 * SEQ * 512) / 256, 256, 0, stream>>>(mo, comb_bf);

  // 11) out = comb @ W_out_bi^T + b_out_bi  (M=2048, N=512, K=512)
  gemm_bf16_wmma<0><<<dim3(512 / 32, 2048 / 64), blk, 0, stream>>>(
      comb_bf, 512, Wobi_bf, 512, b_out_bi, out, nullptr, 512, 512);
}
